// DiffusionModel_11510512353491
// MI455X (gfx1250) — matrix operand order repaired and checked by reading, not compiled, not run on hardware
//
#include <hip/hip_runtime.h>
#include <hip/hip_bf16.h>
#include <math.h>

// ---------------------------------------------------------------------------
// Mamba-style diffusion block for MI455X (gfx1250, wave32, WMMA).
// Heavy GEMMs use v_wmma_f32_16x16x32_bf16 (bf16 in / f32 acc), operands
// converted f32->bf16 while staging tiles into LDS. Global->LDS staging is
// double-buffered (register prefetch + LDS ping-pong, one barrier per tile);
// all fragment ds_loads are hoisted ahead of the WMMA chain so DS latency
// overlaps matrix issue. Scan/conv/LN/head are scalar VALU kernels.
// ---------------------------------------------------------------------------

#define D_MODEL 768
#define D_INNER 1536
#define D_STATE 16
#define D_CONV  4
#define SEQ     1024
#define BATCH   4
#define ROWS    (BATCH * SEQ)        // 4096
#define OUT_DIM (3 * 64 * 64)        // 12288
#define N_LAYERS 4

typedef __bf16 bf16_t;
typedef __attribute__((ext_vector_type(16))) __bf16 v16bf;
typedef __attribute__((ext_vector_type(8)))  __bf16 v8bf;
typedef __attribute__((ext_vector_type(8)))  float  v8f;

__device__ __forceinline__ float siluf(float x) { return x / (1.f + __expf(-x)); }
__device__ __forceinline__ float softplusf(float x) { return (x > 20.f) ? x : log1pf(__expf(x)); }

// ---------------------------------------------------------------------------
// WMMA GEMM: C[M,N] (+)= A[M,K] * B[K,N], A/B f32 row-major, bf16 WMMA math.
// Block: 256 threads = 8 waves; block tile 128x128; K step 32 (1 WMMA K).
// Wave grid 4(M) x 2(N): each wave owns 32x64 = 2x4 fragments of 16x16.
// EPI: 0 = store, 1 = softplus(acc + bias[col]), 2 = C += acc.
// Requires M%128==0, N%128==0, K%32==0 (true for all uses here).
// ---------------------------------------------------------------------------
#define BM 128
#define BN 128
#define BK 32
#define LDA 40   // padded bf16 stride (80 B, keeps 16-B alignment)
#define LDB 40

template <int EPI>
__global__ __launch_bounds__(256) void gemm_wmma(
    const float* __restrict__ A, const float* __restrict__ B,
    float* __restrict__ C, int M, int N, int K, const float* __restrict__ bias)
{
  __shared__ __bf16 As[2][BM * LDA];
  __shared__ __bf16 Bt[2][BN * LDB];   // B tile stored transposed: [n][k]

  const int tid  = threadIdx.x;
  const int lane = tid & 31;
  const int wave = tid >> 5;        // 0..7
  const int wM   = wave & 3;        // 0..3 -> 32 rows each
  const int wN   = wave >> 2;       // 0..1 -> 64 cols each
  const int mBlock = blockIdx.y * BM;
  const int nBlock = blockIdx.x * BN;

  const int halfsel = lane >> 4;    // 0 or 1
  const int lidx    = lane & 15;

  // loader indices
  const int arow = tid >> 3;              // 0..31 (4 passes of 32 rows)
  const int acol = (tid & 7) * 4;         // 0..28
  const int bk   = tid >> 5;              // 0..7  (4 passes of 8 k-rows)
  const int bn   = (tid & 31) * 4;        // 0..124

  v8f acc[2][4];
#pragma unroll
  for (int i = 0; i < 2; ++i)
#pragma unroll
    for (int j = 0; j < 4; ++j)
#pragma unroll
      for (int r = 0; r < 8; ++r) acc[i][j][r] = 0.f;

  float4 av[4], bv[4];   // in-flight global tile (register prefetch)

  auto load_g = [&](int k0) {
#pragma unroll
    for (int p = 0; p < 4; ++p)
      av[p] = *reinterpret_cast<const float4*>(
          A + (size_t)(mBlock + arow + 32 * p) * K + k0 + acol);
#pragma unroll
    for (int p = 0; p < 4; ++p)
      bv[p] = *reinterpret_cast<const float4*>(
          B + (size_t)(k0 + bk + 8 * p) * N + nBlock + bn);
  };

  auto store_l = [&](int buf) {
#pragma unroll
    for (int p = 0; p < 4; ++p) {
      __bf16* dst = &As[buf][(arow + 32 * p) * LDA + acol];
      dst[0] = (__bf16)av[p].x; dst[1] = (__bf16)av[p].y;
      dst[2] = (__bf16)av[p].z; dst[3] = (__bf16)av[p].w;
    }
#pragma unroll
    for (int p = 0; p < 4; ++p) {
      const int k = bk + 8 * p;
      Bt[buf][(bn + 0) * LDB + k] = (__bf16)bv[p].x;
      Bt[buf][(bn + 1) * LDB + k] = (__bf16)bv[p].y;
      Bt[buf][(bn + 2) * LDB + k] = (__bf16)bv[p].z;
      Bt[buf][(bn + 3) * LDB + k] = (__bf16)bv[p].w;
    }
  };

  // prologue: stage tile 0
  load_g(0);
  store_l(0);
  __syncthreads();

  const int nk = K / BK;
  for (int kt = 0; kt < nk; ++kt) {
    const int cur = kt & 1;

    // issue next tile's global loads before computing on this one
    if (kt + 1 < nk) load_g((kt + 1) * BK);

    // ---- hoist ALL fragment ds_loads ahead of the WMMA chain ----
    v16bf afrag[2];
#pragma unroll
    for (int i = 0; i < 2; ++i) {
      const int m = wM * 32 + i * 16 + lidx;
      const v8bf lo = *reinterpret_cast<const v8bf*>(&As[cur][m * LDA + halfsel * 8]);
      const v8bf hi = *reinterpret_cast<const v8bf*>(&As[cur][m * LDA + 16 + halfsel * 8]);
#pragma unroll
      for (int e = 0; e < 8; ++e) { afrag[i][e] = lo[e]; afrag[i][e + 8] = hi[e]; }
    }
    v16bf bfrag[4];
#pragma unroll
    for (int j = 0; j < 4; ++j) {
      const int n = wN * 64 + j * 16 + lidx;
      const v8bf lo = *reinterpret_cast<const v8bf*>(&Bt[cur][n * LDB + halfsel * 8]);
      const v8bf hi = *reinterpret_cast<const v8bf*>(&Bt[cur][n * LDB + 16 + halfsel * 8]);
#pragma unroll
      for (int e = 0; e < 8; ++e) { bfrag[j][e] = lo[e]; bfrag[j][e + 8] = hi[e]; }
    }
    // ---- 8 WMMAs ----
#pragma unroll
    for (int j = 0; j < 4; ++j)
#pragma unroll
      for (int i = 0; i < 2; ++i)
        acc[i][j] = __builtin_amdgcn_wmma_f32_16x16x32_bf16(
            false, afrag[i], false, bfrag[j], (short)0, acc[i][j], false, false);

    // convert + store next tile into the other buffer (waits for loads here)
    if (kt + 1 < nk) store_l(cur ^ 1);
    __syncthreads();
  }

  // ---- epilogue (C/D layout: VGPR r -> M = r + 8*half, N = lane&15) ----
#pragma unroll
  for (int i = 0; i < 2; ++i) {
    const int rowBase = mBlock + wM * 32 + i * 16 + halfsel * 8;
#pragma unroll
    for (int j = 0; j < 4; ++j) {
      const int col = nBlock + wN * 64 + j * 16 + lidx;
#pragma unroll
      for (int r = 0; r < 8; ++r) {
        const size_t idx = (size_t)(rowBase + r) * N + col;
        float v = acc[i][j][r];
        if (EPI == 1) {
          v += bias[col];
          C[idx] = softplusf(v);
        } else if (EPI == 2) {
          C[idx] += v;
        } else {
          C[idx] = v;
        }
      }
    }
  }
}

// ---------------------------------------------------------------------------
// Timestep embedding: emb[b,i] = sin/cos(t_b * freq)
// ---------------------------------------------------------------------------
__global__ void temb_embed_kernel(const int* __restrict__ t, float* __restrict__ emb) {
  const int tid = blockIdx.x * blockDim.x + threadIdx.x;   // 0..3071
  const int b = tid / D_MODEL, i = tid % D_MODEL;
  const float tv = (float)t[b];
  const int half = D_MODEL / 2;
  float out;
  if (i < half) {
    const float f = __expf(-logf(10000.f) * (float)i / (float)half);
    out = __sinf(tv * f);
  } else {
    const float f = __expf(-logf(10000.f) * (float)(i - half) / (float)half);
    out = __cosf(tv * f);
  }
  emb[tid] = out;
}

__global__ void temb_mlp1_kernel(const float* __restrict__ emb,
                                 const float* __restrict__ w1,
                                 const float* __restrict__ b1,
                                 float* __restrict__ hid) {
  const int tid = blockIdx.x * blockDim.x + threadIdx.x;   // 0..12287
  const int b = tid / (4 * D_MODEL), j = tid % (4 * D_MODEL);
  const float* e = emb + (size_t)b * D_MODEL;
  float a = b1[j];
  for (int i = 0; i < D_MODEL; ++i) a += e[i] * w1[(size_t)i * (4 * D_MODEL) + j];
  hid[tid] = siluf(a);
}

__global__ void temb_mlp2_kernel(const float* __restrict__ hid,
                                 const float* __restrict__ w2,
                                 const float* __restrict__ b2,
                                 float* __restrict__ temb) {
  const int tid = blockIdx.x * blockDim.x + threadIdx.x;   // 0..3071
  const int b = tid / D_MODEL, k = tid % D_MODEL;
  const float* hh = hid + (size_t)b * (4 * D_MODEL);
  float a = b2[k];
  for (int j = 0; j < 4 * D_MODEL; ++j) a += hh[j] * w2[(size_t)j * D_MODEL + k];
  temb[tid] = a;
}

__global__ void add_temb_kernel(const float* __restrict__ x,
                                const float* __restrict__ temb,
                                float* __restrict__ h) {
  const size_t tid = (size_t)blockIdx.x * blockDim.x + threadIdx.x;  // rows*768
  const int row = (int)(tid / D_MODEL), d = (int)(tid % D_MODEL);
  const int b = row >> 10;
  h[tid] = x[tid] + temb[b * D_MODEL + d];
}

// ---------------------------------------------------------------------------
// LayerNorm: one block (256 thr) per row of 768
// ---------------------------------------------------------------------------
__global__ __launch_bounds__(256) void ln_kernel(const float* __restrict__ h,
                                                 const float* __restrict__ g,
                                                 const float* __restrict__ b,
                                                 float* __restrict__ z) {
  __shared__ float s1[256], s2[256];
  const int row = blockIdx.x, tid = threadIdx.x;
  const float* hr = h + (size_t)row * D_MODEL;
  float v[3], lsum = 0.f, lsq = 0.f;
#pragma unroll
  for (int i = 0; i < 3; ++i) {
    v[i] = hr[tid + i * 256];
    lsum += v[i]; lsq += v[i] * v[i];
  }
  s1[tid] = lsum; s2[tid] = lsq;
  __syncthreads();
  for (int s = 128; s > 0; s >>= 1) {
    if (tid < s) { s1[tid] += s1[tid + s]; s2[tid] += s2[tid + s]; }
    __syncthreads();
  }
  const float mu = s1[0] * (1.f / D_MODEL);
  const float var = s2[0] * (1.f / D_MODEL) - mu * mu;
  const float rs = rsqrtf(var + 1e-5f);
  float* zr = z + (size_t)row * D_MODEL;
#pragma unroll
  for (int i = 0; i < 3; ++i) {
    const int d = tid + i * 256;
    zr[d] = (v[i] - mu) * rs * g[d] + b[d];
  }
}

// ---------------------------------------------------------------------------
// Causal depthwise conv (K=4) + SiLU. xi = xz[:, :, 0:1536]
// ---------------------------------------------------------------------------
__global__ void conv_silu_kernel(const float* __restrict__ xz,
                                 const float* __restrict__ w,
                                 const float* __restrict__ bias,
                                 float* __restrict__ xc) {
  const size_t tid = (size_t)blockIdx.x * blockDim.x + threadIdx.x;  // rows*1536
  const int row = (int)(tid / D_INNER), c = (int)(tid % D_INNER);
  const int b = row >> 10, l = row & (SEQ - 1);
  float a = bias[c];
#pragma unroll
  for (int j = 0; j < D_CONV; ++j) {
    const int ll = l - (D_CONV - 1) + j;
    if (ll >= 0)
      a += w[c * D_CONV + j] * xz[(size_t)(b * SEQ + ll) * (2 * D_INNER) + c];
  }
  xc[tid] = siluf(a);
}

// ---------------------------------------------------------------------------
// Bst[row, n] = xc[row, :] @ W_x[:, n]   (N = 16, thread per output)
// ---------------------------------------------------------------------------
__global__ void bst_kernel(const float* __restrict__ xc,
                           const float* __restrict__ Wx,
                           float* __restrict__ Bst) {
  const int tid = blockIdx.x * blockDim.x + threadIdx.x;   // rows*16
  const int row = tid >> 4, n = tid & 15;
  const float* xr = xc + (size_t)row * D_INNER;
  float a = 0.f;
  for (int c = 0; c < D_INNER; ++c) a += xr[c] * Wx[c * D_STATE + n];
  Bst[tid] = a;
}

// ---------------------------------------------------------------------------
// Sequential SSM scan over L, one thread per (batch, channel) chain.
// Fuses the gate-SiLU so y is ready for the W_out GEMM.
// ---------------------------------------------------------------------------
__global__ __launch_bounds__(256) void ssm_scan_kernel(
    const float* __restrict__ dt, const float* __restrict__ Bst,
    const float* __restrict__ xc, const float* __restrict__ xz /*gate half*/,
    const float* __restrict__ A_log, const float* __restrict__ Dp,
    float* __restrict__ y) {
  const int tid = blockIdx.x * blockDim.x + threadIdx.x;   // 0..6143
  const int b = tid / D_INNER, c = tid % D_INNER;
  float expA[D_STATE], hs[D_STATE];
#pragma unroll
  for (int n = 0; n < D_STATE; ++n) {
    expA[n] = __expf(A_log[c * D_STATE + n]);
    hs[n] = 0.f;
  }
  const float Dc = Dp[c];
  for (int l = 0; l < SEQ; ++l) {
    const int row = b * SEQ + l;
    const float dtv = dt[(size_t)row * D_INNER + c];
    const float xv  = xc[(size_t)row * D_INNER + c];
    float bl[D_STATE];
#pragma unroll
    for (int n = 0; n < D_STATE; ++n) bl[n] = Bst[(size_t)row * D_STATE + n];
    float a = Dc * xv;
#pragma unroll
    for (int n = 0; n < D_STATE; ++n) {
      hs[n] = hs[n] * __expf(-expA[n] * dtv) + bl[n];
      a += hs[n];
    }
    const float g = xz[(size_t)row * (2 * D_INNER) + D_INNER + c];
    y[(size_t)row * D_INNER + c] = a * siluf(g);
  }
}

// ---------------------------------------------------------------------------
// pooled[b,d] = mean_l h[b,l,d]
// ---------------------------------------------------------------------------
__global__ void pool_kernel(const float* __restrict__ h, float* __restrict__ pooled) {
  const int tid = blockIdx.x * blockDim.x + threadIdx.x;   // 0..3071
  const int b = tid / D_MODEL, d = tid % D_MODEL;
  float a = 0.f;
  for (int l = 0; l < SEQ; ++l) a += h[(size_t)(b * SEQ + l) * D_MODEL + d];
  pooled[tid] = a * (1.f / SEQ);
}

// ---------------------------------------------------------------------------
// out[b,o] = pooled[b,:] @ op_w[:,o] + op_b[o]    (M=4, scalar kernel)
// ---------------------------------------------------------------------------
__global__ void head_kernel(const float* __restrict__ pooled,
                            const float* __restrict__ opw,
                            const float* __restrict__ opb,
                            float* __restrict__ out) {
  const int tid = blockIdx.x * blockDim.x + threadIdx.x;   // 0..49151
  const int b = tid / OUT_DIM, o = tid % OUT_DIM;
  const float* p = pooled + (size_t)b * D_MODEL;
  float a = opb[o];
  for (int i = 0; i < D_MODEL; ++i) a += p[i] * opw[(size_t)i * OUT_DIM + o];
  out[tid] = a;
}

// ---------------------------------------------------------------------------
extern "C" void kernel_launch(void* const* d_in, const int* in_sizes, int n_in,
                              void* d_out, int out_size, void* d_ws, size_t ws_size,
                              hipStream_t stream) {
  const float* x      = (const float*)d_in[0];
  const int*   t      = (const int*)  d_in[1];
  const float* ln_g   = (const float*)d_in[2];
  const float* ln_b   = (const float*)d_in[3];
  const float* W_in   = (const float*)d_in[4];
  const float* conv_w = (const float*)d_in[5];
  const float* conv_b = (const float*)d_in[6];
  const float* A_log  = (const float*)d_in[7];
  const float* Dp     = (const float*)d_in[8];
  const float* W_x    = (const float*)d_in[9];
  const float* W_dt   = (const float*)d_in[10];
  const float* b_dt   = (const float*)d_in[11];
  const float* W_out  = (const float*)d_in[12];
  const float* te_w1  = (const float*)d_in[13];
  const float* te_b1  = (const float*)d_in[14];
  const float* te_w2  = (const float*)d_in[15];
  const float* te_b2  = (const float*)d_in[16];
  const float* op_w   = (const float*)d_in[17];
  const float* op_b   = (const float*)d_in[18];
  float* out = (float*)d_out;

  // workspace carve-up (f32 everywhere), ~152 MB
  float* W = (float*)d_ws;
  size_t o = 0;
  float* h_buf   = W + o; o += (size_t)ROWS * D_MODEL;       // 4096*768
  float* z_buf   = W + o; o += (size_t)ROWS * D_MODEL;
  float* xz_buf  = W + o; o += (size_t)ROWS * 2 * D_INNER;   // 4096*3072
  float* xc_buf  = W + o; o += (size_t)ROWS * D_INNER;
  float* dt_buf  = W + o; o += (size_t)ROWS * D_INNER;
  float* bst_buf = W + o; o += (size_t)ROWS * D_STATE;
  float* y_buf   = W + o; o += (size_t)ROWS * D_INNER;
  float* emb_buf = W + o; o += (size_t)BATCH * D_MODEL;
  float* hid_buf = W + o; o += (size_t)BATCH * 4 * D_MODEL;
  float* tmb_buf = W + o; o += (size_t)BATCH * D_MODEL;
  float* pool_buf= W + o; o += (size_t)BATCH * D_MODEL;

  // --- timestep embedding MLP + h = x + temb ---
  temb_embed_kernel<<<(BATCH * D_MODEL) / 256, 256, 0, stream>>>(t, emb_buf);
  temb_mlp1_kernel<<<(BATCH * 4 * D_MODEL) / 256, 256, 0, stream>>>(emb_buf, te_w1, te_b1, hid_buf);
  temb_mlp2_kernel<<<(BATCH * D_MODEL) / 256, 256, 0, stream>>>(hid_buf, te_w2, te_b2, tmb_buf);
  add_temb_kernel<<<((size_t)ROWS * D_MODEL) / 256, 256, 0, stream>>>(x, tmb_buf, h_buf);

  for (int i = 0; i < N_LAYERS; ++i) {
    const float* Wi  = W_in  + (size_t)i * D_MODEL * 2 * D_INNER;
    const float* Wd  = W_dt  + (size_t)i * D_INNER * D_INNER;
    const float* Wo  = W_out + (size_t)i * D_INNER * D_MODEL;

    ln_kernel<<<ROWS, 256, 0, stream>>>(h_buf, ln_g + i * D_MODEL, ln_b + i * D_MODEL, z_buf);

    // xz = z @ W_in   [4096 x 3072]
    gemm_wmma<0><<<dim3((2 * D_INNER) / BN, ROWS / BM), 256, 0, stream>>>(
        z_buf, Wi, xz_buf, ROWS, 2 * D_INNER, D_MODEL, nullptr);

    // xc = silu(causal dwconv(xi))
    conv_silu_kernel<<<((size_t)ROWS * D_INNER) / 256, 256, 0, stream>>>(
        xz_buf, conv_w + (size_t)i * D_INNER * D_CONV, conv_b + i * D_INNER, xc_buf);

    // dt = softplus(xc @ W_dt + b_dt)   [4096 x 1536]
    gemm_wmma<1><<<dim3(D_INNER / BN, ROWS / BM), 256, 0, stream>>>(
        xc_buf, Wd, dt_buf, ROWS, D_INNER, D_INNER, b_dt + i * D_INNER);

    // Bst = xc @ W_x   [4096 x 16]
    bst_kernel<<<((size_t)ROWS * D_STATE) / 256, 256, 0, stream>>>(
        xc_buf, W_x + (size_t)i * D_INNER * D_STATE, bst_buf);

    // sequential scan + gate-SiLU fusion -> y
    ssm_scan_kernel<<<(BATCH * D_INNER) / 256, 256, 0, stream>>>(
        dt_buf, bst_buf, xc_buf, xz_buf,
        A_log + (size_t)i * D_INNER * D_STATE, Dp + (size_t)i * D_INNER, y_buf);

    // h += y @ W_out   [4096 x 768]
    gemm_wmma<2><<<dim3(D_MODEL / BN, ROWS / BM), 256, 0, stream>>>(
        y_buf, Wo, h_buf, ROWS, D_MODEL, D_INNER, nullptr);
  }

  pool_kernel<<<(BATCH * D_MODEL) / 256, 256, 0, stream>>>(h_buf, pool_buf);
  head_kernel<<<(BATCH * OUT_DIM) / 256, 256, 0, stream>>>(pool_buf, op_w, op_b, out);
}
